// XSAEncoderLayer_7533372637222
// MI455X (gfx1250) — matrix operand order repaired and checked by reading, not compiled, not run on hardware
//
#include <hip/hip_runtime.h>
#include <hip/hip_bf16.h>

typedef __attribute__((ext_vector_type(16))) _Float16 v16h;
typedef __attribute__((ext_vector_type(8)))  _Float16 v8h;
typedef __attribute__((ext_vector_type(8)))  float    v8f;
typedef unsigned int v4u __attribute__((ext_vector_type(4)));
typedef int v4i __attribute__((ext_vector_type(4)));
typedef int v8i_ __attribute__((ext_vector_type(8)));

namespace {
constexpr int B  = 2;
constexpr int C  = 256;
constexpr int NH = 8;
constexpr int HD = 32;
constexpr int T  = 48 * 48;     // 2304
constexpr int TC = T * C;       // 589824
constexpr int C4 = 4 * C;       // 1024
constexpr int NRED = 576;       // TC / 1024 partial blocks per batch
}

#if defined(__has_builtin)
#if __has_builtin(__builtin_amdgcn_tensor_load_to_lds) && \
    __has_builtin(__builtin_amdgcn_s_wait_tensorcnt)
#define HAVE_TDM 1
#endif
#endif
#ifndef HAVE_TDM
#define HAVE_TDM 0
#endif

// ---------------- WMMA fragment helpers (wave32 layouts per CDNA5 ISA 7.12.2) ----

// A-matrix 16x32 f16: lanes 0-15 -> M=lane, halves {K0..7, K16..23};
// lanes 16-31 -> M=lane-16, halves {K8..15, K24..31}.
__device__ __forceinline__ v16h load_a16(const _Float16* base, int lda, int lane) {
  int m  = lane & 15;
  int ks = (lane >> 4) << 3;           // 0 or 8
  const _Float16* p = base + (size_t)m * lda + ks;
  v8h lo = *reinterpret_cast<const v8h*>(p);
  v8h hi = *reinterpret_cast<const v8h*>(p + 16);
  return __builtin_shufflevector(lo, hi, 0,1,2,3,4,5,6,7,8,9,10,11,12,13,14,15);
}

// B-matrix 32x16 f16 taken as W^T of a row-major [N][K] matrix:
// lanes 0-15 -> N=lane, K={0..7, 16..23}; lanes 16-31 -> N=lane-16, K={8..15, 24..31}.
__device__ __forceinline__ v16h load_b16(const _Float16* base, int ldb, int lane) {
  int nn = lane & 15;
  int ks = (lane >> 4) << 3;           // 0 or 8
  const _Float16* p = base + (size_t)nn * ldb + ks;
  v8h lo = *reinterpret_cast<const v8h*>(p);
  v8h hi = *reinterpret_cast<const v8h*>(p + 16);
  return __builtin_shufflevector(lo, hi, 0,1,2,3,4,5,6,7,8,9,10,11,12,13,14,15);
}

__device__ __forceinline__ v8f wmma16(v16h a, v16h b, v8f c) {
  return __builtin_amdgcn_wmma_f32_16x16x32_f16(false, a, false, b, (short)0, c,
                                                false, false);
}

// 16x64 strip: 4 accumulators share one A-fragment per K-step (4 WMMAs / A-load).
__device__ __forceinline__ void gemm_strip4(const _Float16* A, int lda,
                                            const _Float16* W, int ldw,
                                            int row0, int col0, int K, int lane,
                                            v8f acc[4]) {
  const _Float16* ap = A + (size_t)row0 * lda;
  const _Float16* wp = W + (size_t)col0 * ldw;
  for (int k = 0; k < K; k += 32) {
    if (k + 32 < K) __builtin_prefetch(ap + k + 32, 0, 0);   // global_prefetch_b8
    v16h a = load_a16(ap + k, lda, lane);
#pragma unroll
    for (int s = 0; s < 4; ++s) {
      v16h b = load_b16(wp + (size_t)(s * 16) * ldw + k, ldw, lane);
      acc[s] = wmma16(a, b, acc[s]);
    }
  }
}

// ---------------- small utility kernels ------------------------------------------

__global__ __launch_bounds__(256) void cvt_kernel(const float* s, _Float16* d, int n) {
  for (int i = blockIdx.x * 256 + threadIdx.x; i < n; i += gridDim.x * 256)
    d[i] = (_Float16)s[i];
}

// x [B,C,H,W] -> token-major xt16/xt32 [B,T,C]
__global__ __launch_bounds__(256) void pack_kernel(const float* x, _Float16* xt16,
                                                   float* xt32, int total) {
  int idx = blockIdx.x * 256 + threadIdx.x;
  if (idx >= total) return;
  int b = idx / (C * T);
  int rem = idx - b * (C * T);
  int c = rem / T;
  int t = rem - c * T;
  float v = x[idx];
  size_t o = (size_t)b * TC + (size_t)t * C + c;
  xt16[o] = (_Float16)v;
  xt32[o] = v;
}

// ---------------- QKV projection: q/k/v = xt @ W^T + b, stored [B,NH,T,HD] -------

__global__ __launch_bounds__(32) void qkv_kernel(
    const _Float16* xt16, const _Float16* wq16, const _Float16* wk16,
    const _Float16* wv16, const float* bq, const float* bk, const float* bv,
    _Float16* q16, _Float16* k16, _Float16* v16, _Float16* vT16) {
  int lane = threadIdx.x;
  int mt = blockIdx.x >> 2;       // 144 row tiles
  int nt = blockIdx.x & 3;        // 4 strips of 64 columns
  int b  = blockIdx.y;
  int which = blockIdx.z;
  const _Float16* W   = (which == 0) ? wq16 : (which == 1) ? wk16 : wv16;
  const float*  bias  = (which == 0) ? bq   : (which == 1) ? bk   : bv;
  _Float16*     outp  = (which == 0) ? q16  : (which == 1) ? k16  : v16;

  v8f acc[4] = {};
  gemm_strip4(xt16 + (size_t)b * TC, C, W, C, mt * 16, nt * 64, C, lane, acc);

  int n = lane & 15, rofs = (lane >> 4) << 3;
#pragma unroll
  for (int s = 0; s < 4; ++s) {
    int o = nt * 64 + s * 16 + n;
    int h = o >> 5, d = o & 31;
    float bb = bias[o];
#pragma unroll
    for (int j = 0; j < 8; ++j) {
      int t = mt * 16 + rofs + j;
      float v = acc[s][j] + bb;
      outp[((size_t)(b * NH + h) * T + t) * HD + d] = (_Float16)v;
      if (which == 2)
        vT16[((size_t)(b * NH + h) * HD + d) * T + t] = (_Float16)v;
    }
  }
}

// ---------------- attention ------------------------------------------------------
// One wave per (b,h,q-tile). The whole K panel for this (b,h) (T x HD f16,
// 147 KB) is DMA'd into LDS with the Tensor Data Mover; the 16 x T f32 score
// buffer (147 KB) also lives in LDS (320 KB per WGP).

__global__ __launch_bounds__(32) void attn_kernel(
    const _Float16* q16, const _Float16* k16, const _Float16* v16,
    const _Float16* vT16, _Float16* attn16) {
  extern __shared__ float S[];                       // [16][T] scores
  _Float16* Kp = (_Float16*)(S + 16 * T);            // [T][HD] K panel
  float* Otile = (float*)(Kp + (size_t)T * HD);      // [16][32]
  int qt = blockIdx.x;                                // 0..143
  int bh = blockIdx.y;                                // 0..15
  int b = bh >> 3, h = bh & 7;
  int lane = threadIdx.x;

  const _Float16* Qbh = q16 + (size_t)bh * T * HD;
  const _Float16* Kbh = k16 + (size_t)bh * T * HD;
  const _Float16* Vbh = v16 + (size_t)bh * T * HD;
  const _Float16* VT  = vT16 + (size_t)bh * HD * T;

#if HAVE_TDM
  {
    // D# for a 2D tile: 36 rows x 2048 elements of 2 bytes (= T*HD f16, contiguous)
    unsigned lds_addr = (unsigned)(uintptr_t)Kp;     // low 32 bits = LDS offset
    unsigned long long ga = (unsigned long long)(uintptr_t)Kbh;
    const int td0 = 2048, td1 = 36;
    v4u g0 = { 1u,                                    // count=1, user descriptor
               lds_addr,
               (unsigned)ga,
               (unsigned)((ga >> 32) & 0x01FFFFFFu) | (2u << 30) };  // type=2
    v8i_ g1 = { (int)(1u << 16),                      // data_size=1 (2 bytes)
                (int)((unsigned)(td0 & 0xFFFF) << 16),            // tensor_dim0 lo
                (int)(((unsigned)td0 >> 16) | ((unsigned)(td1 & 0xFFFF) << 16)),
                (int)(((unsigned)td1 >> 16) | ((unsigned)td0 << 16)),  // tile_dim0
                (int)td1,                             // tile_dim1 (tile_dim2 = 0)
                (int)td0,                             // tensor_dim0_stride lo
                0, 0 };
    v4i z4 = { 0, 0, 0, 0 };
#if __clang_major__ >= 23
    v8i_ z8 = { 0, 0, 0, 0, 0, 0, 0, 0 };
    __builtin_amdgcn_tensor_load_to_lds(g0, g1, z4, z4, z8, 0);
#else
    __builtin_amdgcn_tensor_load_to_lds(g0, g1, z4, z4, 0);
#endif
    __builtin_amdgcn_s_wait_tensorcnt(0);
  }
#else
  for (int i = lane; i < T * HD / 8; i += 32)
    ((v8h*)Kp)[i] = ((const v8h*)Kbh)[i];
#endif
  __syncthreads();

  const float scale = 0.17677669529663687f;     // 1/sqrt(32)
  int n = lane & 15, rofs = (lane >> 4) << 3;

  // S = scale * Q Kt  (K-dim = HD = 32 -> single WMMA per 16-key tile; K from LDS)
  v16h qa = load_a16(Qbh + (size_t)qt * 16 * HD, HD, lane);
  for (int kt = 0; kt < T / 16; ++kt) {
    v16h kb = load_b16((const _Float16*)Kp + (size_t)kt * 16 * HD, HD, lane);
    v8f s = {};
    s = wmma16(qa, kb, s);
#pragma unroll
    for (int j = 0; j < 8; ++j)
      S[(size_t)(rofs + j) * T + kt * 16 + n] = s[j] * scale;
  }
  __syncthreads();

  // row-wise softmax in LDS (lane r handles query row r)
  if (lane < 16) {
    float* row = S + (size_t)lane * T;
    float m = -3.4e38f;
    for (int k = 0; k < T; ++k) m = fmaxf(m, row[k]);
    float sum = 0.f;
    for (int k = 0; k < T; ++k) { float e = __expf(row[k] - m); row[k] = e; sum += e; }
    float inv = 1.f / sum;
    for (int k = 0; k < T; ++k) row[k] *= inv;
  }
  __syncthreads();

  // O = P V  (A-frag built from LDS fp32 probs; B-frags from transposed V)
  v8f o0 = {}, o1 = {};
  int m_ = lane & 15, ksel = (lane >> 4) << 3;
  for (int kc = 0; kc < T / 32; ++kc) {
    const float* sp = S + (size_t)m_ * T + kc * 32 + ksel;
    v16h pa;
#pragma unroll
    for (int i = 0; i < 8; ++i) pa[i] = (_Float16)sp[i];
#pragma unroll
    for (int i = 0; i < 8; ++i) pa[8 + i] = (_Float16)sp[16 + i];
    v16h b0 = load_b16(VT + kc * 32, T, lane);                   // d = 0..15
    v16h b1 = load_b16(VT + (size_t)16 * T + kc * 32, T, lane);  // d = 16..31
    o0 = wmma16(pa, b0, o0);
    o1 = wmma16(pa, b1, o1);
  }
#pragma unroll
  for (int j = 0; j < 8; ++j) {
    Otile[(rofs + j) * 32 + n]      = o0[j];
    Otile[(rofs + j) * 32 + 16 + n] = o1[j];
  }
  __syncthreads();

  // remove component along normalized v (per query token), write merged heads
  if (lane < 16) {
    int t = qt * 16 + lane;
    float vv[HD];
    float nrm2 = 0.f;
#pragma unroll
    for (int d = 0; d < HD; ++d) {
      vv[d] = (float)Vbh[(size_t)t * HD + d];
      nrm2 += vv[d] * vv[d];
    }
    float inv = 1.f / fmaxf(sqrtf(nrm2), 1e-12f);
    float dot = 0.f;
#pragma unroll
    for (int d = 0; d < HD; ++d) dot += Otile[lane * 32 + d] * vv[d] * inv;
    _Float16* op = attn16 + ((size_t)b * T + t) * C + h * HD;
#pragma unroll
    for (int d = 0; d < HD; ++d)
      op[d] = (_Float16)(Otile[lane * 32 + d] - dot * vv[d] * inv);
  }
}

// ---------------- output projection + residual -----------------------------------

__global__ __launch_bounds__(32) void proj_kernel(
    const _Float16* attn16, const _Float16* wo16, const float* bo,
    const float* xt32, float* r1) {
  int lane = threadIdx.x;
  int mt = blockIdx.x >> 2, nt = blockIdx.x & 3;
  int b = blockIdx.y;
  v8f acc[4] = {};
  gemm_strip4(attn16 + (size_t)b * TC, C, wo16, C, mt * 16, nt * 64, C, lane, acc);
  int n = lane & 15, rofs = (lane >> 4) << 3;
#pragma unroll
  for (int s = 0; s < 4; ++s) {
    int o = nt * 64 + s * 16 + n;
    float bb = bo[o];
#pragma unroll
    for (int j = 0; j < 8; ++j) {
      int t = mt * 16 + rofs + j;
      size_t idx = (size_t)b * TC + (size_t)t * C + o;
      r1[idx] = acc[s][j] + bb + xt32[idx];
    }
  }
}

// ---------------- deterministic GroupNorm(1,C) reductions ------------------------

__global__ __launch_bounds__(256) void reduce_kernel(const float* src, float2* partial) {
  __shared__ float ss[256], sq[256];
  int b = blockIdx.y, blk = blockIdx.x, tid = threadIdx.x;
  const float* p = src + (size_t)b * TC + (size_t)blk * 1024;
  float s = 0.f, q = 0.f;
  for (int i = tid; i < 1024; i += 256) { float v = p[i]; s += v; q += v * v; }
  ss[tid] = s; sq[tid] = q;
  __syncthreads();
  for (int step = 128; step > 0; step >>= 1) {
    if (tid < step) { ss[tid] += ss[tid + step]; sq[tid] += sq[tid + step]; }
    __syncthreads();
  }
  if (tid == 0) partial[b * NRED + blk] = make_float2(ss[0], sq[0]);
}

__global__ __launch_bounds__(256) void finalize_kernel(const float2* partial, float* stats) {
  __shared__ float ss[256], sq[256];
  int b = blockIdx.x, tid = threadIdx.x;
  float s = 0.f, q = 0.f;
  for (int i = tid; i < NRED; i += 256) {
    float2 v = partial[b * NRED + i];
    s += v.x; q += v.y;
  }
  ss[tid] = s; sq[tid] = q;
  __syncthreads();
  for (int step = 128; step > 0; step >>= 1) {
    if (tid < step) { ss[tid] += ss[tid + step]; sq[tid] += sq[tid + step]; }
    __syncthreads();
  }
  if (tid == 0) {
    float mu  = ss[0] / (float)TC;
    float var = sq[0] / (float)TC - mu * mu;
    stats[2 * b]     = mu;
    stats[2 * b + 1] = rsqrtf(var + 1e-5f);
  }
}

__global__ __launch_bounds__(256) void norm1_kernel(
    const float* r1, const float* stats, const float* gamma, const float* beta,
    _Float16* xh, float* xf, int total) {
  int idx = blockIdx.x * 256 + threadIdx.x;
  if (idx >= total) return;
  int b = idx / TC;
  int c = idx % C;
  float v = (r1[idx] - stats[2 * b]) * stats[2 * b + 1] * gamma[c] + beta[c];
  xh[idx] = (_Float16)v;
  xf[idx] = v;
}

__global__ __launch_bounds__(256) void norm2_kernel(
    const float* r2, const float* stats, const float* gamma, const float* beta,
    float* out, int total) {
  int idx = blockIdx.x * 256 + threadIdx.x;
  if (idx >= total) return;
  int b = idx / TC;
  int rem = idx - b * TC;
  int t = rem / C;
  int c = rem - t * C;
  float v = (r2[idx] - stats[2 * b]) * stats[2 * b + 1] * gamma[c] + beta[c];
  out[((size_t)b * C + c) * T + t] = v;     // back to [B,C,H,W]
}

// ---------------- FFN ------------------------------------------------------------

__global__ __launch_bounds__(32) void ffn1_kernel(
    const _Float16* x1h, const _Float16* w1h, const float* bf1, _Float16* hdn) {
  int lane = threadIdx.x;
  int mt = blockIdx.x >> 4, nt = blockIdx.x & 15;     // N = 1024 -> 16 strips
  int b = blockIdx.y;
  v8f acc[4] = {};
  gemm_strip4(x1h + (size_t)b * TC, C, w1h, C, mt * 16, nt * 64, C, lane, acc);
  int n = lane & 15, rofs = (lane >> 4) << 3;
#pragma unroll
  for (int s = 0; s < 4; ++s) {
    int o = nt * 64 + s * 16 + n;
    float bb = bf1[o];
#pragma unroll
    for (int j = 0; j < 8; ++j) {
      int t = mt * 16 + rofs + j;
      hdn[(size_t)b * T * C4 + (size_t)t * C4 + o] = (_Float16)fmaxf(acc[s][j] + bb, 0.f);
    }
  }
}

__global__ __launch_bounds__(32) void ffn2_kernel(
    const _Float16* hdn, const _Float16* w2h, const float* bf2,
    const float* x1f, float* r2) {
  int lane = threadIdx.x;
  int mt = blockIdx.x >> 2, nt = blockIdx.x & 3;
  int b = blockIdx.y;
  v8f acc[4] = {};
  gemm_strip4(hdn + (size_t)b * T * C4, C4, w2h, C4, mt * 16, nt * 64, C4, lane, acc);
  int n = lane & 15, rofs = (lane >> 4) << 3;
#pragma unroll
  for (int s = 0; s < 4; ++s) {
    int o = nt * 64 + s * 16 + n;
    float bb = bf2[o];
#pragma unroll
    for (int j = 0; j < 8; ++j) {
      int t = mt * 16 + rofs + j;
      size_t idx = (size_t)b * TC + (size_t)t * C + o;
      r2[idx] = acc[s][j] + bb + x1f[idx];
    }
  }
}

// ---------------- host side ------------------------------------------------------

extern "C" void kernel_launch(void* const* d_in, const int* in_sizes, int n_in,
                              void* d_out, int out_size, void* d_ws, size_t ws_size,
                              hipStream_t stream) {
  const float* x   = (const float*)d_in[0];
  const float* wq  = (const float*)d_in[1];
  const float* bq  = (const float*)d_in[2];
  const float* wk  = (const float*)d_in[3];
  const float* bk  = (const float*)d_in[4];
  const float* wv  = (const float*)d_in[5];
  const float* bv  = (const float*)d_in[6];
  const float* wo  = (const float*)d_in[7];
  const float* bo  = (const float*)d_in[8];
  const float* g1  = (const float*)d_in[9];
  const float* be1 = (const float*)d_in[10];
  const float* w1  = (const float*)d_in[11];
  const float* bf1 = (const float*)d_in[12];
  const float* w2  = (const float*)d_in[13];
  const float* bf2 = (const float*)d_in[14];
  const float* g2  = (const float*)d_in[15];
  const float* be2 = (const float*)d_in[16];
  float* out = (float*)d_out;

  char* ws = (char*)d_ws;
  size_t off = 0;
  auto alloc = [&](size_t bytes) -> char* {
    char* p = ws + off;
    off += (bytes + 255) & ~(size_t)255;
    return p;
  };
  _Float16* xt16   = (_Float16*)alloc((size_t)B * TC * 2);
  float*    xt32   = (float*)   alloc((size_t)B * TC * 4);
  _Float16* wq16   = (_Float16*)alloc((size_t)C * C * 2);
  _Float16* wk16   = (_Float16*)alloc((size_t)C * C * 2);
  _Float16* wv16   = (_Float16*)alloc((size_t)C * C * 2);
  _Float16* wo16   = (_Float16*)alloc((size_t)C * C * 2);
  _Float16* w1_16  = (_Float16*)alloc((size_t)C4 * C * 2);
  _Float16* w2_16  = (_Float16*)alloc((size_t)C * C4 * 2);
  _Float16* q16    = (_Float16*)alloc((size_t)B * TC * 2);
  _Float16* k16    = (_Float16*)alloc((size_t)B * TC * 2);
  _Float16* v16    = (_Float16*)alloc((size_t)B * TC * 2);
  _Float16* vT16   = (_Float16*)alloc((size_t)B * TC * 2);
  _Float16* attn16 = (_Float16*)alloc((size_t)B * TC * 2);
  float*    r1     = (float*)   alloc((size_t)B * TC * 4);
  _Float16* x1h    = (_Float16*)alloc((size_t)B * TC * 2);
  float*    x1f    = (float*)   alloc((size_t)B * TC * 4);
  _Float16* hdn    = (_Float16*)alloc((size_t)B * T * C4 * 2);
  float*    r2     = (float*)   alloc((size_t)B * TC * 4);
  float2*   partial= (float2*)  alloc((size_t)B * NRED * sizeof(float2));
  float*    stats  = (float*)   alloc((size_t)B * 2 * sizeof(float));

  // weight fp32 -> fp16
  cvt_kernel<<<64, 256, 0, stream>>>(wq, wq16, C * C);
  cvt_kernel<<<64, 256, 0, stream>>>(wk, wk16, C * C);
  cvt_kernel<<<64, 256, 0, stream>>>(wv, wv16, C * C);
  cvt_kernel<<<64, 256, 0, stream>>>(wo, wo16, C * C);
  cvt_kernel<<<256, 256, 0, stream>>>(w1, w1_16, C4 * C);
  cvt_kernel<<<256, 256, 0, stream>>>(w2, w2_16, C * C4);

  // transpose x into token-major fp16/fp32
  pack_kernel<<<(B * C * T + 255) / 256, 256, 0, stream>>>(x, xt16, xt32, B * C * T);

  // QKV projections (WMMA, 16x64 strips)
  qkv_kernel<<<dim3((T / 16) * (C / 64), B, 3), 32, 0, stream>>>(
      xt16, wq16, wk16, wv16, bq, bk, bv, q16, k16, v16, vT16);

  // attention (TDM K-panel -> LDS, WMMA QK^T, LDS softmax, WMMA PV)
  size_t smem = (size_t)(16 * T + 16 * 32) * sizeof(float) + (size_t)T * HD * 2;
  attn_kernel<<<dim3(T / 16, B * NH), 32, smem, stream>>>(q16, k16, v16, vT16, attn16);

  // output projection + residual
  proj_kernel<<<dim3((T / 16) * (C / 64), B), 32, 0, stream>>>(attn16, wo16, bo, xt32, r1);

  // GroupNorm 1
  reduce_kernel<<<dim3(NRED, B), 256, 0, stream>>>(r1, partial);
  finalize_kernel<<<B, 256, 0, stream>>>(partial, stats);
  norm1_kernel<<<(B * TC + 255) / 256, 256, 0, stream>>>(r1, stats, g1, be1, x1h, x1f, B * TC);

  // FFN (WMMA, 16x64 strips)
  ffn1_kernel<<<dim3((T / 16) * (C4 / 64), B), 32, 0, stream>>>(x1h, w1_16, bf1, hdn);
  ffn2_kernel<<<dim3((T / 16) * (C / 64), B), 32, 0, stream>>>(hdn, w2_16, bf2, x1f, r2);

  // GroupNorm 2 -> output [B,C,H,W]
  reduce_kernel<<<dim3(NRED, B), 256, 0, stream>>>(r2, partial);
  finalize_kernel<<<B, 256, 0, stream>>>(partial, stats);
  norm2_kernel<<<(B * TC + 255) / 256, 256, 0, stream>>>(r2, stats, g2, be2, out, B * TC);

  (void)in_sizes; (void)n_in; (void)out_size; (void)ws_size;
}
